// Block_6167573037373
// MI455X (gfx1250) — matrix operand order repaired and checked by reading, not compiled, not run on hardware
//
#include <hip/hip_runtime.h>
#include <hip/hip_bf16.h>
#include <math.h>

// ---------------------------------------------------------------------------
// Equivariant edge message passing for MI455X (gfx1250, wave32, WMMA).
//
// Decomposition: every P-contraction factorizes through per-edge banks
//   bankF  = [xa(64), p=rh.xv(32), q=rh'Xd rh(16), pad(16)]          (128)
//   bankGH[i] = [xv[:,i](32), t_i=Xd.rh(16), s_i=rh'.Xd(16), w_i=rhXxv(32)] (96 each)
//   bankD1[ij] = [xd[:,i,j](16), pad(16)]                            (32 each)
// and six stacked-weight GEMM groups:
//   G1: m_a   (c=64, F=128) over bankF          [P_000 | P_110 | P_220]
//   G2: V1_i  (c=32, F=96 ) over bankGH[i]      [P_011 | P_121 | 0 | P_111]
//   G3: V2    (c=32, F=96 ) over bankF[0:96]    [P_101 | P_211]
//   G4: D1_ij (c=16, F=32 ) over bankD1[ij]     [P_022 | 0]
//   G5: D2_j  (c=16, F=96 ) over bankGH[j]      [P_112 | 0 | P_222 | P_212]
//   G6: D3    (c=16, F=64 ) over bankF[0:64]    [P_202]
// m_v[c,i]   = V1_i[c] + rh_i * V2[c]
// m_d[c,i,j] = D1_ij[c] + rh_i * (D2_j[c] + rh_j * D3[c])
//
// WMMA: A = weights (rows = out-channels), B = bank (cols = 16 edges, LDS).
// rad[l] weighting is a per-lane scalar (edge = column = lane%16):
//   acc += rad[l] * ( sum_kc WMMA(W[l][kc], B[kc]) )
//
// Occupancy: 4 waves/block, 92160 B dynamic LDS -> 3 workgroups (12 waves)
// per 320KB WGP, 3 waves/SIMD to hide the L2 weight-stream latency.
// ---------------------------------------------------------------------------

#define NODES 16384
#define EDGES 262144
#define NTILES (EDGES / 16)
#define OUT_CH 304
#define BANK_STRIDE 720            // bf16 elems per edge in LDS (multiple of 16)
#define WAVES_PER_BLOCK 4
#define NBLOCKS 1024               // 4096 waves -> exactly 4 tiles per wave

// weight group offsets (bf16 elements) inside d_ws
#define OFF_W1 0u                  // 64 x 8 x 128
#define OFF_W2 65536u              // 32 x 8 x 96
#define OFF_W3 90112u              // 32 x 8 x 96
#define OFF_W4 114688u            // 16 x 8 x 32
#define OFF_W5 118784u            // 16 x 8 x 96
#define OFF_W6 131072u            // 16 x 8 x 64
#define W_TOTAL 139264u

typedef __bf16 bf16_t;
typedef __attribute__((ext_vector_type(16))) __bf16 v16bf;
typedef __attribute__((ext_vector_type(8)))  __bf16 v8bf;
typedef __attribute__((ext_vector_type(8)))  float  v8f;

static __device__ inline bf16_t f2bf(float f) {
  union { float f; unsigned u; } x; x.f = f;
  unsigned r = (x.u + 0x7fffu + ((x.u >> 16) & 1u)) >> 16;
  union { unsigned short s; bf16_t b; } y; y.s = (unsigned short)r;
  return y.b;
}

static __device__ inline void gatom(float* p, float v) {
  __hip_atomic_fetch_add(p, v, __ATOMIC_RELAXED, __HIP_MEMORY_SCOPE_AGENT);
}

// B fragment: col = lane&15 (edge), K = kh..kh+15 contiguous from this edge's bank
static __device__ inline v16bf loadB(const bf16_t* bank, int featoff, int lane) {
  const int e2 = lane & 15;
  const int kh = (lane >> 4) << 4;
  return *(const v16bf*)(bank + e2 * BANK_STRIDE + featoff + kh);
}

// One GEMM group for one 16-channel tile: acc[c_tile rows x 16 edges]
// A layout (matches B): lane row = c, elems 0-15 = K[off..off+16),
// off = (lane>=16)*16.
static __device__ inline v8f gemm_group(const bf16_t* __restrict__ Wg, int F, int ct,
                                        int nk, const v16bf* Bfr, const float* radl,
                                        int lane) {
  const int row = lane & 15;
  const int off = (lane >> 4) * 16;
  const bf16_t* wb = Wg + (size_t)(ct * 16 + row) * (size_t)(8 * F) + off;
  v8f acc = {};
  for (int l = 0; l < 8; ++l) {
    v8f part = {};
    const bf16_t* wl = wb + l * F;
    #pragma unroll
    for (int kc = 0; kc < 4; ++kc) {
      if (kc >= nk) break;
      const bf16_t* p = wl + kc * 32;
      v8bf lo = *(const v8bf*)p;
      v8bf hi = *(const v8bf*)(p + 8);
      v16bf a = __builtin_shufflevector(lo, hi, 0,1,2,3,4,5,6,7,8,9,10,11,12,13,14,15);
      part = __builtin_amdgcn_wmma_f32_16x16x32_bf16(false, a, false, Bfr[kc],
                                                     (short)0, part, false, false);
    }
    const float rl = radl[l];
    #pragma unroll
    for (int v = 0; v < 8; ++v) acc[v] += rl * part[v];
  }
  return acc;
}

// ------------------------- weight repack (fp32 -> bf16) ---------------------
__global__ void repack_kernel(const float* __restrict__ P000, const float* __restrict__ P110,
                              const float* __restrict__ P220, const float* __restrict__ P011,
                              const float* __restrict__ P101, const float* __restrict__ P121,
                              const float* __restrict__ P211, const float* __restrict__ P111,
                              const float* __restrict__ P022, const float* __restrict__ P112,
                              const float* __restrict__ P202, const float* __restrict__ P222,
                              const float* __restrict__ P212, bf16_t* __restrict__ W) {
  unsigned idx = blockIdx.x * blockDim.x + threadIdx.x;
  if (idx >= W_TOTAL) return;
  float val = 0.0f;
  unsigned t = idx;
  if (t < OFF_W2) {                       // G1: c64, F128
    unsigned c = t / 1024, l = (t / 128) & 7, k = t & 127, bl = c * 8 + l;
    if (k < 64)        val = P000[bl * 64 + k];
    else if (k < 96)   val = P110[bl * 32 + (k - 64)];
    else if (k < 112)  val = P220[bl * 16 + (k - 96)];
  } else if (t < OFF_W3) {                // G2: c32, F96 over [xv,t,s,w]
    t -= OFF_W2;
    unsigned c = t / 768, l = (t / 96) & 7, k = t % 96, bl = c * 8 + l;
    if (k < 32)        val = P011[bl * 32 + k];
    else if (k < 48)   val = P121[bl * 16 + (k - 32)];
    else if (k >= 64)  val = P111[bl * 32 + (k - 64)];
  } else if (t < OFF_W4) {                // G3: c32, F96 over [xa,p]
    t -= OFF_W3;
    unsigned c = t / 768, l = (t / 96) & 7, k = t % 96, bl = c * 8 + l;
    if (k < 64)        val = P101[bl * 64 + k];
    else               val = P211[bl * 32 + (k - 64)];
  } else if (t < OFF_W5) {                // G4: c16, F32 over xd[:,i,j]
    t -= OFF_W4;
    unsigned c = t / 256, l = (t / 32) & 7, k = t & 31, bl = c * 8 + l;
    if (k < 16)        val = P022[bl * 16 + k];
  } else if (t < OFF_W6) {                // G5: c16, F96 over [xv,t,s,w]
    t -= OFF_W5;
    unsigned c = t / 768, l = (t / 96) & 7, k = t % 96, bl = c * 8 + l;
    if (k < 32)        val = P112[bl * 32 + k];
    else if (k >= 48 && k < 64) val = P222[bl * 16 + (k - 48)];
    else if (k >= 64)  val = P212[bl * 32 + (k - 64)];
  } else {                                // G6: c16, F64 over xa
    t -= OFF_W6;
    unsigned c = t / 512, l = (t / 64) & 7, k = t & 63, bl = c * 8 + l;
    val = P202[bl * 64 + k];
  }
  W[idx] = f2bf(val);
}

__global__ void zero_kernel(float* __restrict__ p, int n) {
  for (int i = blockIdx.x * blockDim.x + threadIdx.x; i < n; i += gridDim.x * blockDim.x)
    p[i] = 0.0f;
}

// ------------------------------ main edge kernel ----------------------------
__global__ __launch_bounds__(WAVES_PER_BLOCK * 32)
void edge_msg_kernel(const float* __restrict__ x_a, const float* __restrict__ x_v,
                     const float* __restrict__ x_d, const float* __restrict__ r_ij,
                     const int* __restrict__ src, const int* __restrict__ dst,
                     const bf16_t* __restrict__ W, float* __restrict__ out) {
  extern __shared__ char smem[];
  const int lane = threadIdx.x & 31;
  const int wave = threadIdx.x >> 5;
  const int gwave = blockIdx.x * WAVES_PER_BLOCK + wave;
  const int nwaves = gridDim.x * WAVES_PER_BLOCK;
  bf16_t* bank = (bf16_t*)smem + (size_t)wave * (16 * BANK_STRIDE);

  const int el = lane & 15;       // this lane's edge within the tile
  const int hi = lane >> 4;       // half-of-wave selector

  for (int tile = gwave; tile < NTILES; tile += nwaves) {
    const int e = tile * 16 + el;

    // ---- per-edge geometry (both halves compute identically for edge el) ----
    const float r0 = r_ij[e * 3 + 0], r1 = r_ij[e * 3 + 1], r2 = r_ij[e * 3 + 2];
    const float xsq = (r0 * r0 + r1 * r1 + r2 * r2) * 0.125f;   // /R0, R0=8
    const float env = fmaxf(1.0f - xsq, 0.0f);
    const float sq  = sqrtf(xsq);
    float radl[8];
    #pragma unroll
    for (int n = 0; n < 8; ++n) radl[n] = cosf(3.14159265358979323f * (float)n * sq) * env;
    const float u0 = r0 * 0.875f, u1 = r1 * 0.875f, u2 = r2 * 0.875f;
    const float un = sqrtf(u0 * u0 + u1 * u1 + u2 * u2);
    const float sc = tanhf(un) / fmaxf(un, 1e-6f);
    float rh[3] = { u0 * sc, u1 * sc, u2 * sc };

    // ---- build bf16 banks in LDS (lane pair (L, L+16) splits edge el) ----
    bf16_t* bk = bank + el * BANK_STRIDE;
    const int s = src[e];
    const float* pa = x_a + (size_t)s * 64;
    for (int f = hi * 32; f < hi * 32 + 32; ++f) bk[f] = f2bf(pa[f]);

    const float* pv = x_v + (size_t)s * 96;
    for (int b = hi * 16; b < hi * 16 + 16; ++b) {
      const float vx = pv[b * 3 + 0], vy = pv[b * 3 + 1], vz = pv[b * 3 + 2];
      bk[64 + b] = f2bf(vx * rh[0] + vy * rh[1] + vz * rh[2]);       // p
      bk[128 + 0 * 96 + b] = f2bf(vx);
      bk[128 + 1 * 96 + b] = f2bf(vy);
      bk[128 + 2 * 96 + b] = f2bf(vz);
      bk[128 + 0 * 96 + 64 + b] = f2bf(rh[1] * vz - rh[2] * vy);     // w = rh x v
      bk[128 + 1 * 96 + 64 + b] = f2bf(rh[2] * vx - rh[0] * vz);
      bk[128 + 2 * 96 + 64 + b] = f2bf(rh[0] * vy - rh[1] * vx);
    }

    const float* pd = x_d + (size_t)s * 144;
    for (int b = hi * 8; b < hi * 8 + 8; ++b) {
      const float m00 = pd[b*9+0], m01 = pd[b*9+1], m02 = pd[b*9+2];
      const float m10 = pd[b*9+3], m11 = pd[b*9+4], m12 = pd[b*9+5];
      const float m20 = pd[b*9+6], m21 = pd[b*9+7], m22 = pd[b*9+8];
      const float t0 = m00*rh[0] + m01*rh[1] + m02*rh[2];            // t = Xd.rh
      const float t1 = m10*rh[0] + m11*rh[1] + m12*rh[2];
      const float t2 = m20*rh[0] + m21*rh[1] + m22*rh[2];
      const float s0 = m00*rh[0] + m10*rh[1] + m20*rh[2];            // s = rh'.Xd
      const float s1 = m01*rh[0] + m11*rh[1] + m21*rh[2];
      const float s2 = m02*rh[0] + m12*rh[1] + m22*rh[2];
      bk[128 + 0*96 + 32 + b] = f2bf(t0);
      bk[128 + 1*96 + 32 + b] = f2bf(t1);
      bk[128 + 2*96 + 32 + b] = f2bf(t2);
      bk[128 + 0*96 + 48 + b] = f2bf(s0);
      bk[128 + 1*96 + 48 + b] = f2bf(s1);
      bk[128 + 2*96 + 48 + b] = f2bf(s2);
      bk[96 + b] = f2bf(rh[0]*t0 + rh[1]*t1 + rh[2]*t2);             // q
      bk[416 + 0*32 + b] = f2bf(m00); bk[416 + 1*32 + b] = f2bf(m01);
      bk[416 + 2*32 + b] = f2bf(m02); bk[416 + 3*32 + b] = f2bf(m10);
      bk[416 + 4*32 + b] = f2bf(m11); bk[416 + 5*32 + b] = f2bf(m12);
      bk[416 + 6*32 + b] = f2bf(m20); bk[416 + 7*32 + b] = f2bf(m21);
      bk[416 + 8*32 + b] = f2bf(m22);
    }
    // zero pads (bankF tail + D1 tails)
    for (int f = 112 + hi * 8; f < 120 + hi * 8; ++f) bk[f] = f2bf(0.0f);
    for (int ij = 0; ij < 9; ++ij)
      for (int f = 16 + hi * 8; f < 24 + hi * 8; ++f) bk[416 + ij * 32 + f] = f2bf(0.0f);

    // intra-wave LDS handoff (cross-lane): drain DS pipe before fragment reads.
    // (LDS ops are in-order within a wave, so this single wait also covers the
    // WAR hazard against the next tile's bank writes.)
    asm volatile("s_wait_dscnt 0" ::: "memory");

    // ---- GEMM stage ----
    v16bf BF[4];
    #pragma unroll
    for (int kc = 0; kc < 4; ++kc) BF[kc] = loadB(bank, kc * 32, lane);

    const int d = dst[e];
    float* ob = out + (size_t)d * OUT_CH;

    // G1: m_a (4 channel tiles)
    for (int ct = 0; ct < 4; ++ct) {
      v8f acc = gemm_group(W + OFF_W1, 128, ct, 4, BF, radl, lane);
      #pragma unroll
      for (int v = 0; v < 8; ++v) gatom(ob + ct * 16 + v + hi * 8, acc[v]);
    }
    // G3: V2 (kept), reuses bankF[0:96]
    v8f V2[2];
    for (int ct = 0; ct < 2; ++ct) V2[ct] = gemm_group(W + OFF_W3, 96, ct, 3, BF, radl, lane);
    // G6: D3 (kept), reuses bankF[0:64]
    v8f D3 = gemm_group(W + OFF_W6, 64, 0, 2, BF, radl, lane);

    // G2: V1_i + combine -> m_v
    for (int i = 0; i < 3; ++i) {
      v16bf BG[4];
      #pragma unroll
      for (int kc = 0; kc < 3; ++kc) BG[kc] = loadB(bank, 128 + i * 96 + kc * 32, lane);
      for (int ct = 0; ct < 2; ++ct) {
        v8f a = gemm_group(W + OFF_W2, 96, ct, 3, BG, radl, lane);
        #pragma unroll
        for (int v = 0; v < 8; ++v) {
          const int c = ct * 16 + v + hi * 8;
          gatom(ob + 64 + c * 3 + i, a[v] + rh[i] * V2[ct][v]);
        }
      }
    }
    // G5: D2_j (kept)
    v8f D2[3];
    for (int j = 0; j < 3; ++j) {
      v16bf BH[4];
      #pragma unroll
      for (int kc = 0; kc < 3; ++kc) BH[kc] = loadB(bank, 128 + j * 96 + kc * 32, lane);
      D2[j] = gemm_group(W + OFF_W5, 96, 0, 3, BH, radl, lane);
    }
    // G4: D1_ij + combine -> m_d
    for (int ij = 0; ij < 9; ++ij) {
      v16bf BD[1];
      BD[0] = loadB(bank, 416 + ij * 32, lane);
      v8f a = gemm_group(W + OFF_W4, 32, 0, 1, BD, radl, lane);
      const int i = ij / 3, j = ij % 3;
      #pragma unroll
      for (int v = 0; v < 8; ++v) {
        const int c = v + hi * 8;
        gatom(ob + 160 + c * 9 + ij, a[v] + rh[i] * (D2[j][v] + rh[j] * D3[v]));
      }
    }
  }
}

// ------------------------------- launcher -----------------------------------
extern "C" void kernel_launch(void* const* d_in, const int* in_sizes, int n_in,
                              void* d_out, int out_size, void* d_ws, size_t ws_size,
                              hipStream_t stream) {
  (void)in_sizes; (void)n_in; (void)ws_size;
  const float* x_a  = (const float*)d_in[0];
  const float* x_v  = (const float*)d_in[1];
  const float* x_d  = (const float*)d_in[2];
  const float* r_ij = (const float*)d_in[3];
  const float* P000 = (const float*)d_in[4];
  const float* P110 = (const float*)d_in[5];
  const float* P220 = (const float*)d_in[6];
  const float* P011 = (const float*)d_in[7];
  const float* P101 = (const float*)d_in[8];
  const float* P121 = (const float*)d_in[9];
  const float* P211 = (const float*)d_in[10];
  const float* P111 = (const float*)d_in[11];
  const float* P022 = (const float*)d_in[12];
  const float* P112 = (const float*)d_in[13];
  const float* P202 = (const float*)d_in[14];
  const float* P222 = (const float*)d_in[15];
  const float* P212 = (const float*)d_in[16];
  const int*   src  = (const int*)d_in[17];
  const int*   dst  = (const int*)d_in[18];
  float* out = (float*)d_out;
  bf16_t* Wws = (bf16_t*)d_ws;

  // 1) repack weights fp32 -> padded bf16 groups
  repack_kernel<<<(W_TOTAL + 255) / 256, 256, 0, stream>>>(
      P000, P110, P220, P011, P101, P121, P211, P111, P022, P112, P202, P222, P212, Wws);

  // 2) zero output (we accumulate with atomics)
  zero_kernel<<<(out_size + 255) / 256, 256, 0, stream>>>(out, out_size);

  // 3) WMMA edge-message kernel: 1024 blocks x 4 waves = 4096 waves,
  //    exactly 4 edge tiles per wave; 92160 B LDS -> 3 workgroups per WGP.
  const size_t shmem = (size_t)WAVES_PER_BLOCK * 16 * BANK_STRIDE * sizeof(bf16_t);
  edge_msg_kernel<<<NBLOCKS, WAVES_PER_BLOCK * 32, shmem, stream>>>(
      x_a, x_v, x_d, r_ij, src, dst, Wws, out);
}
